// BiLevelRoutingAttention_60730837565678
// MI455X (gfx1250) — matrix operand order repaired and checked by reading, not compiled, not run on hardware
//
#include <hip/hip_runtime.h>
#include <hip/hip_bf16.h>
#include <stdint.h>

// ---------------------------------------------------------------------------
// BiLevelRoutingAttention for MI455X (gfx1250, wave32, WMMA 16x16x32 f16)
//   1) f32 -> f16 conversion; x transposed to [b][hw][c] (N-major for GEMM B)
//   2) qkv GEMM  : WMMA, B tile staged to LDS via TDM; each wave computes
//                  2 M-tiles (32x64) so 8 WMMAs amortize one B-tile read
//   3) grid->sequence repack (qs/ks/vs as [b][h][reg][64][32] f16)
//   4) region means + routing top-k
//   5) gathered attention, flash-style online softmax; K chunk token-major,
//      V chunk transposed in LDS so all WMMA fragments are b128 LDS loads
//   6) LePE depthwise 5x5 + bias + add, emits f16 activations N-major
//   7) output GEMM : same WMMA/TDM path
// ---------------------------------------------------------------------------

typedef __attribute__((ext_vector_type(16))) _Float16 v16h;
typedef __attribute__((ext_vector_type(8)))  float    v8f;

union HF16 { v16h v; uint4 q[2]; };   // 32B fragment <-> two 128b loads

#define Bsz   4
#define DIM   512
#define Hh    64
#define Ww    64
#define HW    4096
#define NH    16
#define HD    32
#define NREG  64
#define RS    64
#define TOPK  8

#ifndef USE_TDM
#if defined(__has_builtin)
#  if __has_builtin(__builtin_amdgcn_tensor_load_to_lds) && \
      __has_builtin(__builtin_amdgcn_s_wait_tensorcnt)
#    define USE_TDM 1
#  else
#    define USE_TDM 0
#  endif
#else
#  define USE_TDM 0
#endif
#endif

static __device__ __forceinline__ v8f wmma_f16(const v16h& a, const v16h& b, const v8f& c) {
    return __builtin_amdgcn_wmma_f32_16x16x32_f16(false, a, false, b, (short)0, c, false, false);
}

#if USE_TDM
typedef unsigned int u32x4 __attribute__((ext_vector_type(4)));
typedef int          i32x4 __attribute__((ext_vector_type(4)));
typedef int          i32x8 __attribute__((ext_vector_type(8)));

// 2D tile DMA: global [tensor_d1 x tensor_d0] (2-byte elems, row stride
// stride0 elems) -> LDS tile of tile_d1 rows x tile_d0 elems (contiguous).
// D# layout per CDNA5 ISA.
static __device__ __forceinline__ void tdm_load_tile_2d(
    unsigned int lds_byte_off, unsigned long long gaddr,
    unsigned int tensor_d0, unsigned int tensor_d1,
    unsigned int tile_d0, unsigned int tile_d1, unsigned long long stride0)
{
    u32x4 g0;
    g0[0] = 1u;                                            // count=1 (user D#)
    g0[1] = lds_byte_off;                                  // lds_addr
    g0[2] = (unsigned int)(gaddr & 0xFFFFFFFFu);           // global_addr[31:0]
    g0[3] = (unsigned int)((gaddr >> 32) & 0x01FFFFFFu)    // global_addr[56:32]
          | (2u << 30);                                    // type = 2 (image)
    i32x8 g1;
    g1[0] = (int)(1u << 16);                               // data_size = 1 -> 2B
    g1[1] = (int)((tensor_d0 & 0xFFFFu) << 16);            // tensor_dim0[15:0]
    g1[2] = (int)((tensor_d0 >> 16) | ((tensor_d1 & 0xFFFFu) << 16));
    g1[3] = (int)((tensor_d1 >> 16) | (tile_d0 << 16));    // tile_dim0
    g1[4] = (int)(tile_d1);                                // tile_dim1, dim2=0
    g1[5] = (int)(unsigned int)(stride0 & 0xFFFFFFFFu);    // dim0_stride lo
    g1[6] = (int)(unsigned int)((stride0 >> 32) & 0xFFFFu);
    g1[7] = 0;
    i32x4 g2 = {};
    i32x4 g3 = {};
#if defined(__clang_major__) && (__clang_major__ >= 23)
    i32x8 g4 = {};
    __builtin_amdgcn_tensor_load_to_lds(g0, g1, g2, g3, g4, 0);
#else
    __builtin_amdgcn_tensor_load_to_lds(g0, g1, g2, g3, 0);
#endif
}
#endif

// ---------------- 1a) f32 -> f16 convert ----------------
__global__ void k_f32_to_f16(const float* __restrict__ s, _Float16* __restrict__ d, int n) {
    int i = blockIdx.x * blockDim.x + threadIdx.x;
    if (i < n) d[i] = (_Float16)s[i];
}

// ---------------- 1b) x: [b][c][hw] f32 -> [b][hw][c] f16 (N-major B) -------
__global__ void k_xpose_f32_to_f16(const float* __restrict__ s, _Float16* __restrict__ d) {
    long i = (long)blockIdx.x * blockDim.x + threadIdx.x;   // = ((b*HW)+hw)*DIM+c
    if (i >= (long)Bsz * DIM * HW) return;
    int c  = i & 511;
    int hw = (i >> 9) & 4095;
    int b  = (int)(i >> 21);
    d[i] = (_Float16)s[((long)b * DIM + c) * HW + hw];
}

// ---------------- 2/7) WMMA GEMM ----------------
// C[b][m][n] = sum_k A[m][k] * B[b][n][k] + bias[m]       (B is N-major!)
// Block = 256 threads = 8 waves; block tile 256(M) x 64(N); each wave owns
// 32 M-rows (2 WMMA M-tiles) so one shared B tile feeds 8 WMMAs per K-step.
// B tile (64n x 32k) staged in LDS via TDM double-buffer.
__global__ __launch_bounds__(256) void k_gemm_wmma(
    const _Float16* __restrict__ A,    // [M,K] row major
    const _Float16* __restrict__ Bm,   // [numB,N,K]  (N-major)
    const float*    __restrict__ bias, // [M]
    float*          __restrict__ Cf,   // [numB,M,N] f32 (or null)
    _Float16*       __restrict__ Ch,   // [numB,M,N] f16 (or null)
    int M, int K, int N)
{
    __shared__ __align__(16) _Float16 btile[2][64 * 32];   // [n][k], 2 x 4KB

    int tid  = threadIdx.x;
    int wave = tid >> 5;
    int lane = tid & 31;

    int nTiles  = N >> 6;
    int mBlocks = M >> 8;              // 256 M-rows per block
    int nb = blockIdx.x % nTiles;
    int mb = (blockIdx.x / nTiles) % mBlocks;
    int b  = blockIdx.x / (nTiles * mBlocks);
    int n0 = nb << 6;
    int m0 = (mb << 8) + (wave << 5);  // wave's 32-row slice (2 M-tiles)
    const _Float16* Bp = Bm + (size_t)b * N * K;

    // WMMA lane roles (ISA 7.12.2)
    int am  = lane & 15;           // A: row M
    int akh = (lane >> 4) * 8;     // A: K half-offset
    int bn  = lane & 15;           // B: col N
    int bk  = (lane >> 4) * 8;     // B: K half-offset
    int cn  = lane & 15;           // C/D: col N
    int cm  = (lane >> 4) * 8;     // C/D: row base

    v8f acc[2][4] = {};

#if USE_TDM
    unsigned long long bbase = (unsigned long long)(uintptr_t)Bp;
    unsigned int lds0 = (unsigned int)(uintptr_t)&btile[0][0];
    unsigned int lds1 = (unsigned int)(uintptr_t)&btile[1][0];
    if (wave == 0)   // tile: 64 rows (n) x 32 elems (k), row stride K
        tdm_load_tile_2d(lds0, bbase + (size_t)n0 * K * 2,
                         (unsigned)K, (unsigned)N, 32u, 64u, (unsigned long long)K);
    int cur = 0;
    for (int k0 = 0; k0 < K; k0 += 32, cur ^= 1) {
        if (wave == 0 && (k0 + 32) < K)
            tdm_load_tile_2d(cur ? lds0 : lds1,
                             bbase + ((size_t)n0 * K + (k0 + 32)) * 2,
                             (unsigned)K, (unsigned)N, 32u, 64u, (unsigned long long)K);
        if ((k0 + 32) < K) __builtin_amdgcn_s_wait_tensorcnt(1);
        else               __builtin_amdgcn_s_wait_tensorcnt(0);
        __syncthreads();                           // tile[cur] ready for all waves

        const _Float16* bt = &btile[cur][0];
        HF16 ua0, ua1;
        const _Float16* arow0 = A + (size_t)(m0 + am) * K + k0;
        const _Float16* arow1 = A + (size_t)(m0 + 16 + am) * K + k0;
        ua0.q[0] = *(const uint4*)(arow0 + akh);
        ua0.q[1] = *(const uint4*)(arow0 + 16 + akh);
        ua1.q[0] = *(const uint4*)(arow1 + akh);
        ua1.q[1] = *(const uint4*)(arow1 + 16 + akh);
        #pragma unroll
        for (int t = 0; t < 4; ++t) {
            HF16 ub;
            const _Float16* brow = bt + ((t << 4) + bn) * 32 + bk;   // 16B at k=bk, 16B at k=16+bk
            ub.q[0] = *(const uint4*)(brow);
            ub.q[1] = *(const uint4*)(brow + 16);
            acc[0][t] = wmma_f16(ua0.v, ub.v, acc[0][t]);
            acc[1][t] = wmma_f16(ua1.v, ub.v, acc[1][t]);
        }
        __syncthreads();                           // readers done before re-stage
    }
#else
    for (int k0 = 0; k0 < K; k0 += 32) {
        __syncthreads();
        {   // cooperative coalesced copy: 256 threads x 16B = 4KB tile [n][k]
            int idx = tid * 8;
            int row = idx >> 5, col = idx & 31;
            *(uint4*)&btile[0][idx] =
                *(const uint4*)&Bp[(size_t)(n0 + row) * K + k0 + col];
        }
        __syncthreads();
        const _Float16* bt = &btile[0][0];
        HF16 ua0, ua1;
        const _Float16* arow0 = A + (size_t)(m0 + am) * K + k0;
        const _Float16* arow1 = A + (size_t)(m0 + 16 + am) * K + k0;
        ua0.q[0] = *(const uint4*)(arow0 + akh);
        ua0.q[1] = *(const uint4*)(arow0 + 16 + akh);
        ua1.q[0] = *(const uint4*)(arow1 + akh);
        ua1.q[1] = *(const uint4*)(arow1 + 16 + akh);
        #pragma unroll
        for (int t = 0; t < 4; ++t) {
            HF16 ub;
            const _Float16* brow = bt + ((t << 4) + bn) * 32 + bk;
            ub.q[0] = *(const uint4*)(brow);
            ub.q[1] = *(const uint4*)(brow + 16);
            acc[0][t] = wmma_f16(ua0.v, ub.v, acc[0][t]);
            acc[1][t] = wmma_f16(ua1.v, ub.v, acc[1][t]);
        }
    }
#endif

    size_t obase = (size_t)b * M * N;
    if (Cf) {
        #pragma unroll
        for (int j = 0; j < 2; ++j)
            #pragma unroll
            for (int t = 0; t < 4; ++t)
                #pragma unroll
                for (int r = 0; r < 8; ++r) {
                    int m = m0 + (j << 4) + cm + r;
                    Cf[obase + (size_t)m * N + n0 + (t << 4) + cn] =
                        acc[j][t][r] + bias[m];
                }
    }
    if (Ch) {
        #pragma unroll
        for (int j = 0; j < 2; ++j)
            #pragma unroll
            for (int t = 0; t < 4; ++t)
                #pragma unroll
                for (int r = 0; r < 8; ++r) {
                    int m = m0 + (j << 4) + cm + r;
                    Ch[obase + (size_t)m * N + n0 + (t << 4) + cn] =
                        (_Float16)(acc[j][t][r] + bias[m]);
                }
    }
}

// ---------------- 3) grid -> per-region sequences ----------------
__global__ void k_g2s(const _Float16* __restrict__ qkv,
                      _Float16* __restrict__ qs, _Float16* __restrict__ ks,
                      _Float16* __restrict__ vs)
{
    long i = (long)blockIdx.x * blockDim.x + threadIdx.x;   // B*NH*NREG*RS*HD
    const long total = (long)Bsz * NH * NREG * RS * HD;
    if (i >= total) return;
    int d = i & 31;  long t = i >> 5;
    int p = t & 63;  t >>= 6;
    int reg = t & 63; t >>= 6;
    int h = t & 15;  t >>= 4;
    int b = (int)t;
    int hh = ((reg >> 3) << 3) + (p >> 3);
    int ww = ((reg & 7) << 3) + (p & 7);
    long src = ((long)b * 1536 + h * HD + d) * HW + hh * Ww + ww;
    qs[i] = qkv[src];
    ks[i] = qkv[src + (long)DIM * HW];
    vs[i] = qkv[src + (long)(2 * DIM) * HW];
}

// ---------------- 4a) region means ----------------
__global__ void k_region_means(const _Float16* __restrict__ qkv,
                               float* __restrict__ q_r, float* __restrict__ k_r)
{
    int i = blockIdx.x * blockDim.x + threadIdx.x;   // B*NREG*DIM
    if (i >= Bsz * NREG * DIM) return;
    int c = i & 511;
    int reg = (i >> 9) & 63;
    int b = i >> 15;
    int h0 = (reg >> 3) << 3, w0 = (reg & 7) << 3;
    const _Float16* qb = qkv + ((long)b * 1536 + c) * HW;
    const _Float16* kb = qkv + ((long)b * 1536 + DIM + c) * HW;
    float sq = 0.f, sk = 0.f;
    for (int y = 0; y < 8; ++y)
        for (int x = 0; x < 8; ++x) {
            int hw = (h0 + y) * Ww + (w0 + x);
            sq += (float)qb[hw];
            sk += (float)kb[hw];
        }
    q_r[i] = sq * (1.f / 64.f);
    k_r[i] = sk * (1.f / 64.f);
}

// ---------------- 4b) routing + top-k ----------------
__global__ void k_route_topk(const float* __restrict__ q_r, const float* __restrict__ k_r,
                             int* __restrict__ idx)
{
    int b = blockIdx.x >> 6;
    int qreg = blockIdx.x & 63;
    int j = threadIdx.x;                      // 0..63 candidate region
    __shared__ float sc[NREG];
    const float* qv = q_r + ((long)b * NREG + qreg) * DIM;
    const float* kv = k_r + ((long)b * NREG + j) * DIM;
    float s = 0.f;
    for (int c = 0; c < DIM; ++c) s += qv[c] * kv[c];
    sc[j] = s;
    __syncthreads();
    if (j == 0) {
        #pragma unroll 1
        for (int t = 0; t < TOPK; ++t) {
            int best = 0; float bv = sc[0];
            for (int u = 1; u < NREG; ++u)
                if (sc[u] > bv) { bv = sc[u]; best = u; }
            idx[((long)b * NREG + qreg) * TOPK + t] = best;
            sc[best] = -3.4e38f;
        }
    }
}

// ---------------- 5) gathered attention (flash-style, WMMA) ----------------
// Block = 128 threads = 4 waves, one block per (b, head, region).
// K chunk staged token-major, V chunk staged TRANSPOSED ([d][tok]) so every
// WMMA fragment is two contiguous ds_load_b128.
__global__ __launch_bounds__(128) void k_attention(
    const _Float16* __restrict__ qs, const _Float16* __restrict__ ks,
    const _Float16* __restrict__ vs, const int* __restrict__ idx,
    float* __restrict__ attn_out)
{
    __shared__ __align__(16) _Float16 kbuf[32 * 32];    // [tok][d]  2KB
    __shared__ __align__(16) _Float16 vbufT[32 * 32];   // [d][tok]  2KB
    __shared__ __align__(16) _Float16 psh[4][16 * 32];  // per-wave P relayout

    int blk = blockIdx.x;
    int reg = blk & 63;
    int h   = (blk >> 6) & 15;
    int b   = blk >> 10;
    int w    = threadIdx.x >> 5;
    int lane = threadIdx.x & 31;
    int m0 = w << 4;

    const float scale = 0.044194173824159216f;   // DIM^-0.5

    long headbase = ((long)b * NH + h) * NREG;
    const _Float16* Q = qs + (headbase + reg) * (RS * HD);

    int am  = lane & 15;
    int akh = (lane >> 4) * 8;
    int bn  = lane & 15;
    int bk  = (lane >> 4) * 8;
    int half = lane >> 4;
    int cn  = lane & 15;

    // load Q tile in A layout (16 x 32, K = head dim)
    v16h qa;
    {
        HF16 u;
        const _Float16* qrow = Q + (m0 + am) * HD;
        u.q[0] = *(const uint4*)(qrow + akh);
        u.q[1] = *(const uint4*)(qrow + 16 + akh);
        qa = u.v;
    }

    float mrun[8], lrun[8];
    v8f o0 = {}, o1 = {};
    #pragma unroll
    for (int r = 0; r < 8; ++r) { mrun[r] = -3.4e38f; lrun[r] = 0.f; }

    const int* idxrow = idx + ((long)b * NREG + reg) * TOPK;
    int i8 = threadIdx.x * 8;
    int stok = i8 >> 5;            // staging: token handled by this thread
    int sd0  = i8 & 31;            // staging: first head-dim of its 8 elems

    for (int cc = 0; cc < 16; ++cc) {                 // 16 chunks x 32 tokens = 512
        int kreg = idxrow[cc >> 1];
        long rowbase = (headbase + kreg) * (RS * HD) + (long)(cc & 1) * 32 * HD;

        // prefetch next chunk into cache (data-dependent gather)
        if (cc + 1 < 16) {
            long nb2 = (headbase + idxrow[(cc + 1) >> 1]) * (RS * HD)
                     + (long)((cc + 1) & 1) * 32 * HD;
            __builtin_prefetch(&ks[nb2 + i8], 0, 1);
            __builtin_prefetch(&vs[nb2 + i8], 0, 1);
        }

        __syncthreads();                              // previous chunk consumed
        {   // cooperative stage: K token-major copy, V transposed scatter
            *(uint4*)&kbuf[i8] = *(const uint4*)&ks[rowbase + i8];
            union { uint4 q; _Float16 hh[8]; } uv;
            uv.q = *(const uint4*)&vs[rowbase + i8];
            #pragma unroll
            for (int j = 0; j < 8; ++j)
                vbufT[(sd0 + j) * 32 + stok] = uv.hh[j];
        }
        __syncthreads();

        // ---- S = Q K^T : two 16x16 C tiles, single K-step of 32
        v16h kb0, kb1;
        {
            HF16 u0, u1;
            const _Float16* k0p = &kbuf[bn * HD + bk];
            const _Float16* k1p = &kbuf[(16 + bn) * HD + bk];
            u0.q[0] = *(const uint4*)(k0p);  u0.q[1] = *(const uint4*)(k0p + 16);
            u1.q[0] = *(const uint4*)(k1p);  u1.q[1] = *(const uint4*)(k1p + 16);
            kb0 = u0.v; kb1 = u1.v;
        }
        v8f s0 = {}, s1 = {};
        s0 = wmma_f16(qa, kb0, s0);
        s1 = wmma_f16(qa, kb1, s1);

        // ---- online softmax update (rows replicated across 16-lane halves)
        #pragma unroll
        for (int r = 0; r < 8; ++r) {
            float a0 = s0[r] * scale;
            float a1 = s1[r] * scale;
            float v = fmaxf(a0, a1);
            v = fmaxf(v, __shfl_xor(v, 1, 16));
            v = fmaxf(v, __shfl_xor(v, 2, 16));
            v = fmaxf(v, __shfl_xor(v, 4, 16));
            v = fmaxf(v, __shfl_xor(v, 8, 16));
            float mn   = fmaxf(mrun[r], v);
            float corr = __expf(mrun[r] - mn);
            float p0 = __expf(a0 - mn);
            float p1 = __expf(a1 - mn);
            float ps = p0 + p1;
            ps += __shfl_xor(ps, 1, 16);
            ps += __shfl_xor(ps, 2, 16);
            ps += __shfl_xor(ps, 4, 16);
            ps += __shfl_xor(ps, 8, 16);
            lrun[r] = lrun[r] * corr + ps;
            mrun[r] = mn;
            o0[r] *= corr;
            o1[r] *= corr;
            int row = half * 8 + r;
            psh[w][row * 32 + cn]      = (_Float16)p0;
            psh[w][row * 32 + 16 + cn] = (_Float16)p1;
        }

        // ---- reload P in A layout (K = token-in-chunk)
        v16h pa;
        {
            HF16 u;
            const _Float16* pr = &psh[w][am * 32];
            u.q[0] = *(const uint4*)(pr + akh);
            u.q[1] = *(const uint4*)(pr + 16 + akh);
            pa = u.v;
        }

        // ---- O += P * V : V^T in LDS -> contiguous fragments
        v16h vb0, vb1;
        {
            HF16 u0, u1;
            const _Float16* v0p = &vbufT[cn * 32 + bk];         // hd col cn
            const _Float16* v1p = &vbufT[(16 + cn) * 32 + bk];  // hd col 16+cn
            u0.q[0] = *(const uint4*)(v0p);  u0.q[1] = *(const uint4*)(v0p + 16);
            u1.q[0] = *(const uint4*)(v1p);  u1.q[1] = *(const uint4*)(v1p + 16);
            vb0 = u0.v; vb1 = u1.v;
        }
        o0 = wmma_f16(pa, vb0, o0);
        o1 = wmma_f16(pa, vb1, o1);
    }

    // ---- normalize and scatter back to spatial layout [b][c][h][w]
    #pragma unroll
    for (int r = 0; r < 8; ++r) {
        int p = m0 + half * 8 + r;
        float inv = 1.f / lrun[r];
        int hh = ((reg >> 3) << 3) + (p >> 3);
        int ww = ((reg & 7) << 3) + (p & 7);
        long base = ((long)b * DIM + h * HD) * HW + hh * Ww + ww;
        attn_out[base + (long)cn * HW]        = o0[r] * inv;
        attn_out[base + (long)(16 + cn) * HW] = o1[r] * inv;
    }
}

// ---------------- 6) LePE depthwise 5x5 + biases -> f16 N-major ----------------
__global__ void k_lepe(const _Float16* __restrict__ qkv,
                       const float* __restrict__ lepe_w, const float* __restrict__ lepe_b,
                       const float* __restrict__ attn, _Float16* __restrict__ y_h)
{
    long i = (long)blockIdx.x * blockDim.x + threadIdx.x;    // B*DIM*H*W
    if (i >= (long)Bsz * DIM * HW) return;
    int x  = i & 63;
    int yv = (i >> 6) & 63;
    int c  = (i >> 12) & 511;
    int b  = (int)(i >> 21);
    const _Float16* v = qkv + ((long)b * 1536 + 2 * DIM + c) * HW;
    const float* wk = lepe_w + c * 25;
    float acc = 0.f;
    #pragma unroll
    for (int ky = 0; ky < 5; ++ky) {
        int yy = yv + ky - 2;
        if (yy < 0 || yy > 63) continue;
        #pragma unroll
        for (int kx = 0; kx < 5; ++kx) {
            int xx = x + kx - 2;
            if (xx < 0 || xx > 63) continue;
            acc += wk[ky * 5 + kx] * (float)v[yy * Ww + xx];
        }
    }
    // write N-major ([b][hw][c]) so output GEMM's B tile is contiguous per lane
    y_h[((long)b * HW + (yv * Ww + x)) * DIM + c] =
        (_Float16)(attn[i] + acc + lepe_b[c]);
}

// ---------------------------------------------------------------------------
extern "C" void kernel_launch(void* const* d_in, const int* in_sizes, int n_in,
                              void* d_out, int out_size, void* d_ws, size_t ws_size,
                              hipStream_t stream) {
    const float* x      = (const float*)d_in[0];
    const float* qkv_w  = (const float*)d_in[1];
    const float* qkv_b  = (const float*)d_in[2];
    const float* lepe_w = (const float*)d_in[3];
    const float* lepe_b = (const float*)d_in[4];
    const float* out_w  = (const float*)d_in[5];
    const float* out_b  = (const float*)d_in[6];
    float* out = (float*)d_out;

    // workspace carve (all sizes 256B multiples), ~171 MB total
    char* p = (char*)d_ws;
    size_t off = 0;
    auto carve = [&](size_t bytes) { char* q = p + off; off += (bytes + 255) & ~(size_t)255; return q; };
    _Float16* x_t    = (_Float16*)carve((size_t)Bsz * DIM * HW * 2);     // [b][hw][c]
    _Float16* wqkv_h = (_Float16*)carve((size_t)3 * DIM * DIM * 2);
    _Float16* wout_h = (_Float16*)carve((size_t)DIM * DIM * 2);
    _Float16* qkv_h  = (_Float16*)carve((size_t)Bsz * 3 * DIM * HW * 2); // [b][o][hw]
    _Float16* qs     = (_Float16*)carve((size_t)Bsz * DIM * HW * 2);
    _Float16* ks     = (_Float16*)carve((size_t)Bsz * DIM * HW * 2);
    _Float16* vs     = (_Float16*)carve((size_t)Bsz * DIM * HW * 2);
    float*    q_r    = (float*)carve((size_t)Bsz * NREG * DIM * 4);
    float*    k_r    = (float*)carve((size_t)Bsz * NREG * DIM * 4);
    int*      idxb   = (int*)carve((size_t)Bsz * NREG * TOPK * 4);
    float*    attn_o = (float*)carve((size_t)Bsz * DIM * HW * 4);
    _Float16* y_h    = (_Float16*)carve((size_t)Bsz * DIM * HW * 2);     // [b][hw][c]
    (void)ws_size; (void)in_sizes; (void)n_in; (void)out_size;

    // 1) precision conversion (+ transpose of x to N-major)
    {
        long n = (long)Bsz * DIM * HW;
        k_xpose_f32_to_f16<<<(int)((n + 255) / 256), 256, 0, stream>>>(x, x_t);
        int m = 3 * DIM * DIM;
        k_f32_to_f16<<<(m + 255) / 256, 256, 0, stream>>>(qkv_w, wqkv_h, m);
        m = DIM * DIM;
        k_f32_to_f16<<<(m + 255) / 256, 256, 0, stream>>>(out_w, wout_h, m);
    }

    // 2) qkv GEMM: blocks = B * (1536/256) * (4096/64) = 1536
    k_gemm_wmma<<<1536, 256, 0, stream>>>(wqkv_h, x_t, qkv_b, nullptr, qkv_h,
                                          3 * DIM, DIM, HW);

    // 3) grid -> sequences
    {
        long n = (long)Bsz * NH * NREG * RS * HD;   // 8.39M
        k_g2s<<<(int)((n + 255) / 256), 256, 0, stream>>>(qkv_h, qs, ks, vs);
    }

    // 4) routing
    k_region_means<<<(Bsz * NREG * DIM + 255) / 256, 256, 0, stream>>>(qkv_h, q_r, k_r);
    k_route_topk<<<Bsz * NREG, 64, 0, stream>>>(q_r, k_r, idxb);

    // 5) attention: one block per (b, head, region)
    k_attention<<<Bsz * NH * NREG, 128, 0, stream>>>(qs, ks, vs, idxb, attn_o);

    // 6) LePE + bias + residual add (emits N-major f16)
    {
        long n = (long)Bsz * DIM * HW;
        k_lepe<<<(int)((n + 255) / 256), 256, 0, stream>>>(qkv_h, lepe_w, lepe_b, attn_o, y_h);
    }

    // 7) output GEMM: blocks = B * (512/256) * (4096/64) = 512
    k_gemm_wmma<<<512, 256, 0, stream>>>(wout_h, y_h, out_b, out, nullptr,
                                         DIM, DIM, HW);
}
